// SSMGate3_14224931684841
// MI455X (gfx1250) — matrix operand order repaired and checked by reading, not compiled, not run on hardware
//
#include <hip/hip_runtime.h>
#include <hip/hip_bf16.h>
#include <math.h>

// ---------------------------------------------------------------------------
// MI455X (gfx1250) implementation of the 4-layer RoPE-SSM reference.
//  - GEMMs: v_wmma_f32_16x16x32_bf16, 128x128 tiles, double-buffered LDS,
//           async global->LDS staging (ASYNCcnt) when the builtin exists.
//  - Weights pre-transposed to N x K bf16 once per layer so both A and B
//    tiles stage as contiguous b128 copies (no in-loop transpose).
//  - Scan: 64 blocks (b,h), state in registers, LDS staging per step.
// ---------------------------------------------------------------------------

typedef __attribute__((ext_vector_type(16))) __bf16 v16bf;
typedef __attribute__((ext_vector_type(8)))  float  v8f;
typedef __bf16 bf16_t;
typedef int v4i __attribute__((vector_size(16)));

union FragU { uint4 u[2]; v16bf v; };

// model constants
#define CB   4
#define CL   1024
#define CW   1024
#define CH   16
#define CN   64
#define CNH  32
#define CE   2048
#define CP   128
#define CBL  (CB * CL)
#define CNF  4
#define CNT  20

// GEMM tiling
#define BM  128
#define BN  128
#define KT  32
#define KTP 40   // padded LDS row (bf16 elems); 80B rows keep 16B alignment

#if defined(__gfx1250__) && defined(__has_builtin)
#if __has_builtin(__builtin_amdgcn_global_load_async_to_lds_b128) && \
    __has_builtin(__builtin_amdgcn_s_wait_asynccnt)
#define USE_ASYNC_LDS 1
#endif
#endif
#ifndef USE_ASYNC_LDS
#define USE_ASYNC_LDS 0
#endif

// builtin expects pointers to 16-byte int vectors (global / LDS addr spaces)
#define GPTR(p) ((__attribute__((address_space(1))) v4i*)(p))
#define LPTR(p) ((__attribute__((address_space(3))) v4i*)(p))

// ---------------------------------------------------------------------------
// fp32 (K x N) -> bf16 (N x K) tiled transpose-cast for weights
// ---------------------------------------------------------------------------
__global__ __launch_bounds__(256) void transpose_cast_kernel(
    const float* __restrict__ in, bf16_t* __restrict__ out, int K, int N) {
  __shared__ float tile[32][33];
  const int bk = blockIdx.x * 32;
  const int bn = blockIdx.y * 32;
  const int tx = threadIdx.x & 31;
  const int ty = threadIdx.x >> 5;           // 8 row-groups
  for (int r = ty; r < 32; r += 8)
    tile[r][tx] = in[(size_t)(bk + r) * N + bn + tx];
  __syncthreads();
  for (int r = ty; r < 32; r += 8)
    out[(size_t)(bn + r) * K + bk + tx] = (bf16_t)tile[tx][r];
}

// ---------------------------------------------------------------------------
// RMSNorm: one block per row of W=1024. Optional bf16 and fp32 outputs.
// ---------------------------------------------------------------------------
__global__ __launch_bounds__(256) void rmsnorm_kernel(
    const float* __restrict__ x, const float* __restrict__ w,
    bf16_t* __restrict__ out_bf, float* __restrict__ out_f) {
  const int row = blockIdx.x;
  const int t = threadIdx.x;
  const float* xr = x + (size_t)row * CW;
  float ss = 0.f;
  for (int i = t; i < CW; i += 256) { float v = xr[i]; ss += v * v; }
  __shared__ float red[256];
  red[t] = ss;
  __syncthreads();
  for (int s = 128; s > 0; s >>= 1) {
    if (t < s) red[t] += red[t + s];
    __syncthreads();
  }
  const float scale = rsqrtf(red[0] * (1.0f / CW) + 1e-6f);
  for (int i = t; i < CW; i += 256) {
    float v = xr[i] * scale * w[i];
    if (out_bf) out_bf[(size_t)row * CW + i] = (bf16_t)v;
    if (out_f)  out_f[(size_t)row * CW + i] = v;
  }
}

// ---------------------------------------------------------------------------
// bf16 WMMA GEMM:  C[M,N] = A[M,K] @ Bt[N,K]^T + bias (+ residual)
// A row-major, Bt = transposed weights (N x K). 256 threads = 8 waves,
// wave tile 32x64 = 2x4 WMMA 16x16 tiles. Double-buffered LDS staging.
// ---------------------------------------------------------------------------
__global__ __launch_bounds__(256) void gemm_bf16_wmma_kernel(
    const bf16_t* __restrict__ A, const bf16_t* __restrict__ Bt,
    const float* __restrict__ bias, const float* __restrict__ resid,
    float* __restrict__ C, int M, int N, int K) {
  __shared__ bf16_t As[2][BM * KTP];
  __shared__ bf16_t Bs[2][BN * KTP];

  const int tid  = threadIdx.x;
  const int lane = tid & 31;
  const int wid  = tid >> 5;
  const int wm   = (wid >> 1) * 32;
  const int wn   = (wid & 1) * 64;
  const int bm0  = blockIdx.y * BM;
  const int bn0  = blockIdx.x * BN;

  const bf16_t* gA = A  + (size_t)bm0 * K;
  const bf16_t* gB = Bt + (size_t)bn0 * K;

  // per-thread staging offsets: v in 0..511 -> row v>>2, 8-elem chunk (v&3)*8
  const int v0 = tid * 2, v1 = tid * 2 + 1;
  const int m0s = v0 >> 2, kv0 = (v0 & 3) * 8;
  const int m1s = v1 >> 2, kv1 = (v1 & 3) * 8;

  auto stage = [&](int kt, int buf) {
    const int kb = kt * KT;
#if USE_ASYNC_LDS
    __builtin_amdgcn_global_load_async_to_lds_b128(
        GPTR(gA + (size_t)m0s * K + kb + kv0), LPTR(&As[buf][m0s * KTP + kv0]), 0, 0);
    __builtin_amdgcn_global_load_async_to_lds_b128(
        GPTR(gA + (size_t)m1s * K + kb + kv1), LPTR(&As[buf][m1s * KTP + kv1]), 0, 0);
    __builtin_amdgcn_global_load_async_to_lds_b128(
        GPTR(gB + (size_t)m0s * K + kb + kv0), LPTR(&Bs[buf][m0s * KTP + kv0]), 0, 0);
    __builtin_amdgcn_global_load_async_to_lds_b128(
        GPTR(gB + (size_t)m1s * K + kb + kv1), LPTR(&Bs[buf][m1s * KTP + kv1]), 0, 0);
#else
    uint4 a0 = *(const uint4*)(gA + (size_t)m0s * K + kb + kv0);
    uint4 a1 = *(const uint4*)(gA + (size_t)m1s * K + kb + kv1);
    uint4 b0 = *(const uint4*)(gB + (size_t)m0s * K + kb + kv0);
    uint4 b1 = *(const uint4*)(gB + (size_t)m1s * K + kb + kv1);
    *(uint4*)(&As[buf][m0s * KTP + kv0]) = a0;
    *(uint4*)(&As[buf][m1s * KTP + kv1]) = a1;
    *(uint4*)(&Bs[buf][m0s * KTP + kv0]) = b0;
    *(uint4*)(&Bs[buf][m1s * KTP + kv1]) = b1;
#endif
  };

  v8f acc[2][4];
#pragma unroll
  for (int i = 0; i < 2; ++i)
#pragma unroll
    for (int j = 0; j < 4; ++j)
      acc[i][j] = (v8f){0.f, 0.f, 0.f, 0.f, 0.f, 0.f, 0.f, 0.f};

  const int nk = K / KT;
  stage(0, 0);

  for (int kt = 0; kt < nk; ++kt) {
    const int buf = kt & 1;
#if USE_ASYNC_LDS
    __builtin_amdgcn_s_wait_asynccnt(0);
#endif
    __syncthreads();
    if (kt + 1 < nk) stage(kt + 1, buf ^ 1);
    if (kt + 2 < nk) {   // keep L2 warm two tiles ahead
      __builtin_prefetch(gA + (size_t)(tid >> 1) * K + (kt + 2) * KT, 0, 0);
      __builtin_prefetch(gB + (size_t)(tid >> 1) * K + (kt + 2) * KT, 0, 0);
    }

    // A fragments: lane<16 -> K{0..7,16..23}; lane>=16 -> +8
    v16bf af[2];
    {
      const int rA = lane & 15;
      const int kb = (lane >> 4) * 8;
#pragma unroll
      for (int i = 0; i < 2; ++i) {
        const bf16_t* base = &As[buf][(wm + i * 16 + rA) * KTP];
        FragU f;
        f.u[0] = *(const uint4*)(base + kb);
        f.u[1] = *(const uint4*)(base + 16 + kb);
        af[i] = f.v;
      }
    }
    // B fragments: lane<16 -> K{0..7,16..23}; lane>=16 -> +8 (n-major LDS)
    v16bf bfr[4];
    {
      const int rN  = lane & 15;
      const int kb2 = (lane >> 4) * 8;
#pragma unroll
      for (int j = 0; j < 4; ++j) {
        const bf16_t* base = &Bs[buf][(wn + j * 16 + rN) * KTP + kb2];
        FragU f;
        f.u[0] = *(const uint4*)(base);
        f.u[1] = *(const uint4*)(base + 16);
        bfr[j] = f.v;
      }
    }
#pragma unroll
    for (int i = 0; i < 2; ++i)
#pragma unroll
      for (int j = 0; j < 4; ++j)
        acc[i][j] = __builtin_amdgcn_wmma_f32_16x16x32_bf16(
            false, af[i], false, bfr[j], (short)0, acc[i][j], false, false);
  }

  // writeback: C frag lane -> (N = lane&15, M base = 8*(lane>>4))
  const int cn_l   = lane & 15;
  const int cm_off = (lane >> 4) * 8;
#pragma unroll
  for (int i = 0; i < 2; ++i) {
#pragma unroll
    for (int j = 0; j < 4; ++j) {
      const int gm = bm0 + wm + i * 16 + cm_off;
      const int gn = bn0 + wn + j * 16 + cn_l;
      const float bns = bias ? bias[gn] : 0.f;
#pragma unroll
      for (int r = 0; r < 8; ++r) {
        size_t off = (size_t)(gm + r) * N + gn;
        float v = acc[i][j][r] + bns;
        if (resid) v += resid[off];
        C[off] = v;
      }
    }
  }
}

// ---------------------------------------------------------------------------
// prep: dt = softplus(r@dt_w + dt_b)*keep, decay = exp(dt*A), cos/sin tables,
//       x_in *= keep (in place on xz[:, :E]). One block per (b,l) row.
// ---------------------------------------------------------------------------
__global__ __launch_bounds__(256) void prep_kernel(
    const bf16_t* __restrict__ r, const float* __restrict__ dtw,
    const float* __restrict__ dtb, const float* __restrict__ Alog,
    const float* __restrict__ omega, const unsigned char* __restrict__ mask,
    float* __restrict__ xz, float* __restrict__ dt_o,
    float* __restrict__ dec_o, float* __restrict__ cos_o,
    float* __restrict__ sin_o) {
  const int row = blockIdx.x;
  const int t = threadIdx.x;
  const int hh = t & 15, seg = t >> 4;
  const bf16_t* rr = r + (size_t)row * CW;

  float acc = 0.f;
  const int k0 = seg * 64;
  for (int k = k0; k < k0 + 64; ++k) acc += (float)rr[k] * dtw[k * CH + hh];

  __shared__ float red[256];
  __shared__ float dtL[CH];
  red[t] = acc;
  __syncthreads();

  const float keep = mask[row] ? 0.f : 1.f;
  if (t < CH) {
    float s = 0.f;
    for (int g = 0; g < 16; ++g) s += red[g * 16 + t];
    s += dtb[t];
    float sp = (s > 20.f) ? s : log1pf(expf(s));   // softplus
    sp *= keep;
    float Av = -expf(Alog[t]);
    dt_o[(size_t)row * CH + t]  = sp;
    dec_o[(size_t)row * CH + t] = expf(sp * Av);
    dtL[t] = sp;
  }
  __syncthreads();

  for (int idx = t; idx < CH * CNH; idx += 256) {
    int h2 = idx >> 5, j = idx & 31;
    float sv, cv;
    __sincosf(dtL[h2] * omega[j], &sv, &cv);
    cos_o[(size_t)row * (CH * CNH) + idx] = cv;
    sin_o[(size_t)row * (CH * CNH) + idx] = sv;
  }
  for (int e = t; e < CE; e += 256) xz[(size_t)row * (2 * CE) + e] *= keep;
}

// ---------------------------------------------------------------------------
// Sequential scan over L. Block = one (b,h); thread t owns p = t>>1 and
// 16 RoPE pairs (n, n+32) with n in [16*(t&1), 16*(t&1)+16).
// ---------------------------------------------------------------------------
__global__ __launch_bounds__(256) void scan_kernel(
    const float* __restrict__ xz, const float* __restrict__ bc,
    const float* __restrict__ dt, const float* __restrict__ dec,
    const float* __restrict__ cosb, const float* __restrict__ sinb,
    float* __restrict__ ys) {
  const int b = blockIdx.x >> 4;
  const int h = blockIdx.x & 15;
  const int t = threadIdx.x;
  const int p = t >> 1;
  const int half = t & 1;
  const int nbase = half * 16;

  __shared__ float BtL[CN], CtL[CN], cL[CNH], sL[CNH], sc[2];

  float hs[32], us[32];
#pragma unroll
  for (int i = 0; i < 32; ++i) { hs[i] = 0.f; us[i] = 0.f; }

  for (int l = 0; l < CL; ++l) {
    const size_t row = (size_t)b * CL + l;
    __syncthreads();                       // previous-step LDS reads done
    if (t < 64)        BtL[t]       = bc[row * (2*CH*CN) + h * CN + t];
    else if (t < 128)  CtL[t - 64]  = bc[row * (2*CH*CN) + CH*CN + h * CN + (t - 64)];
    else if (t < 160)  cL[t - 128]  = cosb[(row * CH + h) * CNH + (t - 128)];
    else if (t < 192)  sL[t - 160]  = sinb[(row * CH + h) * CNH + (t - 160)];
    else if (t == 192) sc[0]        = dt[row * CH + h];
    else if (t == 193) sc[1]        = dec[row * CH + h];
    __syncthreads();

    const float xp  = xz[row * (2 * CE) + h * CP + p];   // masked x_in
    const float hdt = sc[0] * 0.5f;
    const float dv  = sc[1];
    float yp = 0.f;
#pragma unroll
    for (int j = 0; j < 16; ++j) {
      const int n = nbase + j;
      const float c = cL[n], s = sL[n];
      const float h1 = hs[j], h2 = hs[j + 16];
      const float u1 = us[j], u2 = us[j + 16];
      const float rh1 = h1 * c - h2 * s, rh2 = h1 * s + h2 * c;
      const float ru1 = u1 * c - u2 * s, ru2 = u1 * s + u2 * c;
      const float un1 = xp * BtL[n], un2 = xp * BtL[n + 32];
      const float hn1 = dv * rh1 + hdt * (dv * ru1 + un1);
      const float hn2 = dv * rh2 + hdt * (dv * ru2 + un2);
      yp += hn1 * CtL[n] + hn2 * CtL[n + 32];
      hs[j] = hn1; hs[j + 16] = hn2;
      us[j] = un1; us[j + 16] = un2;
    }
    const float ysum = yp + __shfl_down(yp, 1);          // pair-reduce over n
    if (half == 0) ys[row * CE + h * CP + p] = ysum;
  }
}

// ---------------------------------------------------------------------------
// gate: yg = (ys + x_in * D_skip[h]) * silu(z)  -> bf16 for out-proj GEMM
// ---------------------------------------------------------------------------
__global__ __launch_bounds__(256) void gate_kernel(
    const float* __restrict__ ys, const float* __restrict__ xz,
    const float* __restrict__ D, bf16_t* __restrict__ yg, size_t total) {
  for (size_t idx = (size_t)blockIdx.x * 256 + threadIdx.x; idx < total;
       idx += (size_t)gridDim.x * 256) {
    const size_t row = idx >> 11;            // / E
    const int e = (int)(idx & (CE - 1));
    const int hh = e >> 7;                   // / P
    const float y = ys[idx] + xz[row * (2 * CE) + e] * D[hh];
    const float z = xz[row * (2 * CE) + CE + e];
    const float sg = 1.f / (1.f + expf(-z));
    yg[idx] = (bf16_t)(y * z * sg);
  }
}

// ---------------------------------------------------------------------------
// head: attention pooling over L + two tiny projections. Block per batch b.
// out layout: [features (B,NF)] then [biases (B,NT)] flattened.
// ---------------------------------------------------------------------------
__global__ __launch_bounds__(256) void head_kernel(
    const float* __restrict__ xn, const unsigned char* __restrict__ mask,
    const float* __restrict__ q, const float* __restrict__ fw,
    const float* __restrict__ fb, const float* __restrict__ bw,
    const float* __restrict__ bb, float* __restrict__ out) {
  const int b = blockIdx.x;
  const int t = threadIdx.x;
  __shared__ float qs[CW];
  __shared__ float a[CL];
  __shared__ float red[256];
  __shared__ float pooled[CW];

  for (int i = t; i < CW; i += 256) qs[i] = q[i];
  __syncthreads();

  for (int l = t; l < CL; l += 256) {
    const float* xr = xn + ((size_t)b * CL + l) * CW;
    float s = 0.f;
    for (int k = 0; k < CW; ++k) s += qs[k] * xr[k];
    s *= 0.03125f;                          // W0^-0.5
    if (mask[b * CL + l]) s = -1e9f;
    a[l] = s;
  }
  __syncthreads();

  float m = -1e30f;
  for (int l = t; l < CL; l += 256) m = fmaxf(m, a[l]);
  red[t] = m;
  __syncthreads();
  for (int s2 = 128; s2 > 0; s2 >>= 1) {
    if (t < s2) red[t] = fmaxf(red[t], red[t + s2]);
    __syncthreads();
  }
  m = red[0];
  __syncthreads();

  float ssum = 0.f;
  for (int l = t; l < CL; l += 256) {
    float e = expf(a[l] - m);
    a[l] = e;
    ssum += e;
  }
  red[t] = ssum;
  __syncthreads();
  for (int s2 = 128; s2 > 0; s2 >>= 1) {
    if (t < s2) red[t] += red[t + s2];
    __syncthreads();
  }
  const float inv = 1.f / red[0];
  __syncthreads();

  for (int wv = t; wv < CW; wv += 256) {
    float s = 0.f;
    for (int l = 0; l < CL; ++l) s += a[l] * xn[((size_t)b * CL + l) * CW + wv];
    pooled[wv] = s * inv;
  }
  __syncthreads();

  if (t < CNF) {
    float s = fb[t];
    for (int k = 0; k < CW; ++k) s += pooled[k] * fw[k * CNF + t];
    out[b * CNF + t] = s;
  } else if (t < CNF + CNT) {
    const int j = t - CNF;
    float s = bb[j];
    for (int k = 0; k < CW; ++k) s += pooled[k] * bw[k * CNT + j];
    out[CB * CNF + b * CNT + j] = s;
  }
}

// ---------------------------------------------------------------------------
// host orchestration
// ---------------------------------------------------------------------------
extern "C" void kernel_launch(void* const* d_in, const int* in_sizes, int n_in,
                              void* d_out, int out_size, void* d_ws,
                              size_t ws_size, hipStream_t stream) {
  (void)in_sizes; (void)n_in; (void)out_size; (void)ws_size;
  const float* h_in   = (const float*)d_in[0];
  const unsigned char* mask = (const unsigned char*)d_in[1];
  const float* ln_w   = (const float*)d_in[2];
  const float* in_w   = (const float*)d_in[3];
  const float* in_b   = (const float*)d_in[4];
  const float* dt_w   = (const float*)d_in[5];
  const float* dt_b   = (const float*)d_in[6];
  const float* bc_w   = (const float*)d_in[7];
  const float* bc_b   = (const float*)d_in[8];
  const float* A_log  = (const float*)d_in[9];
  const float* omega  = (const float*)d_in[10];
  const float* D_skip = (const float*)d_in[11];
  const float* out_w  = (const float*)d_in[12];
  const float* out_b  = (const float*)d_in[13];
  const float* fin_ln = (const float*)d_in[14];
  const float* quer   = (const float*)d_in[15];
  const float* feat_w = (const float*)d_in[16];
  const float* feat_b = (const float*)d_in[17];
  const float* bias_w = (const float*)d_in[18];
  const float* bias_b = (const float*)d_in[19];
  float* outp = (float*)d_out;

  char* ws = (char*)d_ws;
  size_t off = 0;
  auto alloc = [&](size_t bytes) -> void* {
    void* p = ws + off;
    off += (bytes + 255) & ~(size_t)255;
    return p;
  };
  float*  x     = (float*)alloc((size_t)CBL * CW * 4);
  bf16_t* rbf   = (bf16_t*)alloc((size_t)CBL * CW * 2);
  float*  xz    = (float*)alloc((size_t)CBL * 2 * CE * 4);
  float*  bc    = (float*)alloc((size_t)CBL * 2 * CH * CN * 4);
  float*  dtb_s = (float*)alloc((size_t)CBL * CH * 4);
  float*  dec_s = (float*)alloc((size_t)CBL * CH * 4);
  float*  cos_s = (float*)alloc((size_t)CBL * CH * CNH * 4);
  float*  sin_s = (float*)alloc((size_t)CBL * CH * CNH * 4);
  float*  ys    = (float*)alloc((size_t)CBL * CE * 4);
  bf16_t* yg    = (bf16_t*)alloc((size_t)CBL * CE * 2);
  bf16_t* win   = (bf16_t*)alloc((size_t)CW * 2 * CE * 2);   // (2E x W) bf16
  bf16_t* wbc   = (bf16_t*)alloc((size_t)CW * 2 * CH * CN * 2);
  bf16_t* wout  = (bf16_t*)alloc((size_t)CE * CW * 2);
  float*  xn    = (float*)alloc((size_t)CBL * CW * 4);

  (void)hipMemcpyAsync(x, h_in, (size_t)CBL * CW * 4,
                       hipMemcpyDeviceToDevice, stream);

  for (int d = 0; d < 4; ++d) {
    const float* ln_d  = ln_w  + (size_t)d * CW;
    const float* inw_d = in_w  + (size_t)d * CW * 2 * CE;
    const float* inb_d = in_b  + (size_t)d * 2 * CE;
    const float* dtw_d = dt_w  + (size_t)d * CW * CH;
    const float* dtb_d = dt_b  + (size_t)d * CH;
    const float* bcw_d = bc_w  + (size_t)d * CW * 2 * CH * CN;
    const float* bcb_d = bc_b  + (size_t)d * 2 * CH * CN;
    const float* Al_d  = A_log + (size_t)d * CH;
    const float* om_d  = omega + (size_t)d * CNH;
    const float* Ds_d  = D_skip+ (size_t)d * CH;
    const float* ow_d  = out_w + (size_t)d * CE * CW;
    const float* ob_d  = out_b + (size_t)d * CW;

    // weights: fp32 (K x N) -> bf16 (N x K)
    transpose_cast_kernel<<<dim3(CW / 32, 2 * CE / 32), 256, 0, stream>>>(
        inw_d, win, CW, 2 * CE);
    transpose_cast_kernel<<<dim3(CW / 32, 2 * CH * CN / 32), 256, 0, stream>>>(
        bcw_d, wbc, CW, 2 * CH * CN);
    transpose_cast_kernel<<<dim3(CE / 32, CW / 32), 256, 0, stream>>>(
        ow_d, wout, CE, CW);

    rmsnorm_kernel<<<CBL, 256, 0, stream>>>(x, ln_d, rbf, nullptr);

    // xz = r @ in_w + in_b   (4096 x 4096, K=1024)
    gemm_bf16_wmma_kernel<<<dim3(2 * CE / BN, CBL / BM), 256, 0, stream>>>(
        rbf, win, inb_d, nullptr, xz, CBL, 2 * CE, CW);
    // bc = r @ bc_w + bc_b   (4096 x 2048, K=1024)
    gemm_bf16_wmma_kernel<<<dim3(2 * CH * CN / BN, CBL / BM), 256, 0, stream>>>(
        rbf, wbc, bcb_d, nullptr, bc, CBL, 2 * CH * CN, CW);

    prep_kernel<<<CBL, 256, 0, stream>>>(rbf, dtw_d, dtb_d, Al_d, om_d, mask,
                                         xz, dtb_s, dec_s, cos_s, sin_s);
    scan_kernel<<<CB * CH, 256, 0, stream>>>(xz, bc, dtb_s, dec_s, cos_s,
                                             sin_s, ys);
    gate_kernel<<<8192, 256, 0, stream>>>(ys, xz, Ds_d, yg, (size_t)CBL * CE);

    // x = x + yg @ out_w + out_b   (4096 x 1024, K=2048), in-place residual
    gemm_bf16_wmma_kernel<<<dim3(CW / BN, CBL / BM), 256, 0, stream>>>(
        yg, wout, ob_d, x, x, CBL, CW, CE);
  }

  rmsnorm_kernel<<<CBL, 256, 0, stream>>>(x, fin_ln, nullptr, xn);
  head_kernel<<<CB, 256, 0, stream>>>(xn, mask, quer, feat_w, feat_b, bias_w,
                                      bias_b, outp);
}
